// GraphSage_51900384805180
// MI455X (gfx1250) — matrix operand order repaired and checked by reading, not compiled, not run on hardware
//
#include <hip/hip_runtime.h>
#include <stdint.h>

typedef __attribute__((ext_vector_type(16))) __bf16 v16bf;
typedef __attribute__((ext_vector_type(8)))  __bf16 v8bf;
typedef __attribute__((ext_vector_type(8)))  float  v8f;

#define D_IN   256
#define D_OUT  128
#define K_NB   10
#define N1     40960
#define BATCH  4096

#define XS1 (2 * D_IN + 16)   // 528 ushorts per LDS row (padded)
#define XS2 (2 * D_OUT + 16)  // 272 ushorts per LDS row (padded)

static __device__ __forceinline__ unsigned short f2bf(float f) {
    union { float f; unsigned u; } v; v.f = f;
    unsigned r = v.u + 0x7FFFu + ((v.u >> 16) & 1u);   // round-to-nearest-even
    return (unsigned short)(r >> 16);
}
static __device__ __forceinline__ float bf2f(unsigned short s) {
    union { unsigned u; float f; } v; v.u = ((unsigned)s) << 16;
    return v.f;
}

// ---------------- kernel 0: fp32 -> bf16 weight conversion (runs once per launch,
// results stay L2-resident across the GEMM kernels) ----------------
__global__ void __launch_bounds__(256)
wconv_kernel(const float* __restrict__ W1, const float* __restrict__ W2,
             unsigned short* __restrict__ W1b, unsigned short* __restrict__ W2b) {
    int i = blockIdx.x * 256 + threadIdx.x;
    if (i < D_OUT * 2 * D_IN)  W1b[i] = f2bf(W1[i]);
    if (i < D_OUT * 2 * D_OUT) W2b[i] = f2bf(W2[i]);
}

// ---------------- kernel 1: layer 1 ----------------
// Per block: 16 rows of X=[self|agg] (bf16, LDS), then 8 waves x 16 WMMA k-steps.
__global__ void __launch_bounds__(256)
sage_layer1_kernel(const float* __restrict__ raw,
                   const int* __restrict__ nodes1,
                   const int* __restrict__ neighs1,
                   const unsigned short* __restrict__ W1b,
                   unsigned short* __restrict__ h1b) {
    __shared__ __align__(16) unsigned short Xs[16 * XS1];
    const int tid = threadIdx.x;                 // 256 threads == D_IN columns
    const size_t row0 = (size_t)blockIdx.x * 16;

    // coalesced gather: each thread owns one feature column for all 16 rows
    for (int r = 0; r < 16; ++r) {
        const int self = nodes1[row0 + r];
        Xs[r * XS1 + tid] = f2bf(raw[(size_t)self * D_IN + tid]);
        const int* nb = neighs1 + (row0 + r) * K_NB;
        float acc = 0.f;
        #pragma unroll
        for (int j = 0; j < K_NB; ++j)
            acc += raw[(size_t)nb[j] * D_IN + tid];
        Xs[r * XS1 + D_IN + tid] = f2bf(acc * 0.1f);
    }
    __syncthreads();

    const int lane = tid & 31;
    const int wv   = tid >> 5;        // n-tile 0..7 (8 waves cover 128 outputs)
    const int n    = lane & 15;
    const int hi   = lane >> 4;
    const int kA   = hi << 4;         // A-fragment k base within 32-chunk: 0 or 16
    const int kB   = hi << 4;         // B-fragment k base within 32-chunk: 0 or 16

    const unsigned short* wrow = W1b + (size_t)(wv * 16 + n) * (2 * D_IN) + kB;
    const unsigned short* arow = Xs + n * XS1 + kA;   // A: m = lane & 15

    v8f c = {};
    #pragma unroll
    for (int ks = 0; ks < 16; ++ks) {            // K = 512 = 16 x 32
        union { v16bf v; v8bf h[2]; } a;
        a.h[0] = *(const v8bf*)(arow + ks * 32);       // K = k0 .. k0+7
        a.h[1] = *(const v8bf*)(arow + ks * 32 + 8);   // K = k0+8 .. k0+15
        v16bf b = *(const v16bf*)(wrow + ks * 32);     // 32 contiguous bytes of W row
        c = __builtin_amdgcn_wmma_f32_16x16x32_bf16(false, a.v, false, b,
                                                    (short)0, c, false, false);
    }

    #pragma unroll
    for (int r = 0; r < 8; ++r) {                // C/D layout: M = r + 8*hi, N = lane&15
        const int m = r + (hi << 3);
        float v = c[r];
        v = v > 0.f ? v : 0.f;
        h1b[(row0 + m) * D_OUT + wv * 16 + n] = f2bf(v);
    }
}

// ---------------- kernel 2: layer 2 ----------------
__global__ void __launch_bounds__(256)
sage_layer2_kernel(const unsigned short* __restrict__ h1b,
                   const int* __restrict__ map2,
                   const int* __restrict__ neighs2,
                   const unsigned short* __restrict__ W2b,
                   float* __restrict__ out) {
    __shared__ __align__(16) unsigned short Xs[16 * XS2];
    const int tid = threadIdx.x;
    const size_t row0 = (size_t)blockIdx.x * 16;

    const int col = tid & 127;                    // D_OUT = 128 columns
    for (int r = tid >> 7; r < 16; r += 2) {      // 2 row-groups in parallel
        const int self = map2[row0 + r];
        Xs[r * XS2 + col] = h1b[(size_t)self * D_OUT + col];
        const int* nb = neighs2 + (row0 + r) * K_NB;
        float acc = 0.f;
        #pragma unroll
        for (int j = 0; j < K_NB; ++j)
            acc += bf2f(h1b[(size_t)nb[j] * D_OUT + col]);
        Xs[r * XS2 + D_OUT + col] = f2bf(acc * 0.1f);
    }
    __syncthreads();

    const int lane = tid & 31;
    const int wv   = tid >> 5;
    const int n    = lane & 15;
    const int hi   = lane >> 4;
    const int kA   = hi << 4;
    const int kB   = hi << 4;

    const unsigned short* wrow = W2b + (size_t)(wv * 16 + n) * (2 * D_OUT) + kB;
    const unsigned short* arow = Xs + n * XS2 + kA;

    v8f c = {};
    #pragma unroll
    for (int ks = 0; ks < 8; ++ks) {             // K = 256 = 8 x 32
        union { v16bf v; v8bf h[2]; } a;
        a.h[0] = *(const v8bf*)(arow + ks * 32);
        a.h[1] = *(const v8bf*)(arow + ks * 32 + 8);
        v16bf b = *(const v16bf*)(wrow + ks * 32);
        c = __builtin_amdgcn_wmma_f32_16x16x32_bf16(false, a.v, false, b,
                                                    (short)0, c, false, false);
    }

    #pragma unroll
    for (int r = 0; r < 8; ++r) {
        const int m = r + (hi << 3);
        float v = c[r];
        out[(row0 + m) * D_OUT + wv * 16 + n] = v > 0.f ? v : 0.f;
    }
}

extern "C" void kernel_launch(void* const* d_in, const int* in_sizes, int n_in,
                              void* d_out, int out_size, void* d_ws, size_t ws_size,
                              hipStream_t stream) {
    const float* raw    = (const float*)d_in[0];
    const float* W1     = (const float*)d_in[1];
    const float* W2     = (const float*)d_in[2];
    const int*   nodes1 = (const int*)d_in[3];
    const int*   neighs1= (const int*)d_in[4];
    const int*   map2   = (const int*)d_in[5];
    const int*   neighs2= (const int*)d_in[6];
    float* out = (float*)d_out;

    // workspace layout (bf16): h1 [N1 x 128], W1 [128 x 512], W2 [128 x 256]
    unsigned short* h1b = (unsigned short*)d_ws;
    unsigned short* W1b = h1b + (size_t)N1 * D_OUT;
    unsigned short* W2b = W1b + (size_t)D_OUT * 2 * D_IN;

    wconv_kernel<<<256, 256, 0, stream>>>(W1, W2, W1b, W2b);
    sage_layer1_kernel<<<N1 / 16, 256, 0, stream>>>(raw, nodes1, neighs1, W1b, h1b);
    sage_layer2_kernel<<<BATCH / 16, 256, 0, stream>>>(h1b, map2, neighs2, W2b, out);
}
